// CapsuleLayer_69002944578106
// MI455X (gfx1250) — matrix operand order repaired and checked by reading, not compiled, not run on hardware
//
#include <hip/hip_runtime.h>

// CapsuleLayer fused kernel for MI455X (gfx1250, wave32).
// GEMM (16384x2048 @ 2048x160) via V_WMMA_F32_16X16X32_BF16 with hi/lo bf16
// split (3 WMMAs per K-step, fp32 accumulate), fused dynamic-routing epilogue.
// Fragments are loaded in native WMMA layout: A from LDS via ds_load_b128,
// B from a pre-split K-major bf16 copy of W via global_load_b128.

#define IN_DIM 2048
#define NC 10
#define DC 16
#define NCOL (NC * DC)     // 160
#define KCH 64             // K-chunk staged in LDS
#define ASTR 72            // padded A row stride in bf16 units (36 dwords)

typedef __attribute__((ext_vector_type(16))) __bf16 v16bf;
typedef __attribute__((ext_vector_type(8)))  float  v8f;

struct uint4x2 { uint4 a, b; };

__device__ __forceinline__ unsigned short f32_to_bf16_rne(float f) {
    unsigned int u = __builtin_bit_cast(unsigned int, f);
    unsigned int r = u + 0x7FFFu + ((u >> 16) & 1u);
    return (unsigned short)(r >> 16);
}
__device__ __forceinline__ float bf16_bits_to_f32(unsigned short h) {
    return __builtin_bit_cast(float, ((unsigned int)h) << 16);
}

__device__ __forceinline__ v16bf make_frag(uint4 q0, uint4 q1) {
    uint4x2 t{q0, q1};
    return __builtin_bit_cast(v16bf, t);
}

__device__ __forceinline__ v8f wmma_bf16(v16bf a, v16bf b, v8f c) {
    return __builtin_amdgcn_wmma_f32_16x16x32_bf16(
        /*neg_a=*/false, a, /*neg_b=*/false, b,
        /*c_mod=*/(short)0, c, /*reuse_a=*/false, /*reuse_b=*/false);
}

// ---- prologue: split W (fp32 [2048][160]) into K-major bf16 hi/lo [160][2048]
__global__ __launch_bounds__(256)
void split_w_kernel(const float* __restrict__ W,
                    unsigned short* __restrict__ wt_hi,
                    unsigned short* __restrict__ wt_lo) {
    int t = blockIdx.x * 256 + threadIdx.x;      // 0 .. 2048*160-1
    if (t >= IN_DIM * NCOL) return;
    int k   = t / NCOL;
    int col = t - k * NCOL;
    float wv = W[t];
    unsigned short h = f32_to_bf16_rne(wv);
    wt_hi[col * IN_DIM + k] = h;
    wt_lo[col * IN_DIM + k] = f32_to_bf16_rne(wv - bf16_bits_to_f32(h));
}

__global__ __launch_bounds__(320)
void capsule_fused_kernel(const float* __restrict__ A,
                          const unsigned short* __restrict__ wt_hi,
                          const unsigned short* __restrict__ wt_lo,
                          float* __restrict__ out) {
    __shared__ unsigned short lds_ahi[2][16 * ASTR];  // bf16 hi, fragment-layout rows
    __shared__ unsigned short lds_alo[2][16 * ASTR];  // bf16 lo
    __shared__ float lds_u[16 * NC * DC];             // inputs_hat for this strip

    const int tid  = threadIdx.x;
    const int wave = tid >> 5;          // 0..9 -> capsule index
    const int lane = tid & 31;
    const int d    = lane & 15;
    const int half = lane >> 4;
    const int row0 = blockIdx.x * 16;
    const int arow = lane & 15;         // A-fragment M row
    const int colg = wave * DC + d;     // global column in [0,160)

    // A-chunk loader mapping (threads 0..255): 16 rows x 16 float4 each
    const int lrow = tid >> 4;
    const int lkq  = tid & 15;

    v8f acc = {0.f, 0.f, 0.f, 0.f, 0.f, 0.f, 0.f, 0.f};

    // B-fragment base pointers (K-major bf16; elems i -> K = base + i)
    const unsigned short* bh_base = wt_hi + (size_t)colg * IN_DIM + 8 * half;
    const unsigned short* bl_base = wt_lo + (size_t)colg * IN_DIM + 8 * half;

    // prologue: prefetch first A chunk
    float4 staged;
    if (tid < 256) {
        staged = *reinterpret_cast<const float4*>(
            A + (size_t)(row0 + lrow) * IN_DIM + lkq * 4);
    }

    int buf = 0;
    for (int kc = 0; kc < IN_DIM; kc += KCH) {
        // ---- convert + store staged A chunk into LDS (fragment layout) ----
        if (tid < 256) {
            unsigned short h0 = f32_to_bf16_rne(staged.x);
            unsigned short h1 = f32_to_bf16_rne(staged.y);
            unsigned short h2 = f32_to_bf16_rne(staged.z);
            unsigned short h3 = f32_to_bf16_rne(staged.w);
            unsigned short l0 = f32_to_bf16_rne(staged.x - bf16_bits_to_f32(h0));
            unsigned short l1 = f32_to_bf16_rne(staged.y - bf16_bits_to_f32(h1));
            unsigned short l2 = f32_to_bf16_rne(staged.z - bf16_bits_to_f32(h2));
            unsigned short l3 = f32_to_bf16_rne(staged.w - bf16_bits_to_f32(h3));
            uint2 hv, lv;
            hv.x = (unsigned int)h0 | ((unsigned int)h1 << 16);
            hv.y = (unsigned int)h2 | ((unsigned int)h3 << 16);
            lv.x = (unsigned int)l0 | ((unsigned int)l1 << 16);
            lv.y = (unsigned int)l2 | ((unsigned int)l3 << 16);
            *reinterpret_cast<uint2*>(&lds_ahi[buf][lrow * ASTR + lkq * 4]) = hv;
            *reinterpret_cast<uint2*>(&lds_alo[buf][lrow * ASTR + lkq * 4]) = lv;
        }
        __syncthreads();
        // prefetch next chunk while this one is consumed
        if (tid < 256 && (kc + KCH) < IN_DIM) {
            staged = *reinterpret_cast<const float4*>(
                A + (size_t)(row0 + lrow) * IN_DIM + (kc + KCH) + lkq * 4);
        }

        // ---- two K=32 WMMA steps per chunk ----
        #pragma unroll
        for (int ks = 0; ks < KCH; ks += 32) {
            // A fragments: elems 0..7 -> K=8h+0..7, elems 8..15 -> K=16+8h+0..7
            const unsigned short* ah = &lds_ahi[buf][arow * ASTR + ks + 8 * half];
            const unsigned short* al = &lds_alo[buf][arow * ASTR + ks + 8 * half];
            v16bf ahi = make_frag(*reinterpret_cast<const uint4*>(ah),
                                  *reinterpret_cast<const uint4*>(ah + 16));
            v16bf alo = make_frag(*reinterpret_cast<const uint4*>(al),
                                  *reinterpret_cast<const uint4*>(al + 16));

            // B fragments: elems 0..7 -> K=8h+0..7, elems 8..15 -> K=16+8h+0..7
            const uint4* bh = reinterpret_cast<const uint4*>(bh_base + kc + ks);
            const uint4* bl = reinterpret_cast<const uint4*>(bl_base + kc + ks);
            v16bf bhi = make_frag(bh[0], bh[2]);
            v16bf blo = make_frag(bl[0], bl[2]);

            // bf16x3 split: hi*hi + hi*lo + lo*hi  (fp32 accumulate)
            acc = wmma_bf16(ahi, bhi, acc);
            acc = wmma_bf16(ahi, blo, acc);
            acc = wmma_bf16(alo, bhi, acc);
        }
        buf ^= 1;
    }

    // ---- spill inputs_hat tiles to LDS: u[row][capsule][dim] ----
    #pragma unroll
    for (int r = 0; r < 8; ++r) {
        int row = r + 8 * half;                  // C layout: VGPR r -> M = r + 8*half
        lds_u[(row * NC + wave) * DC + d] = acc[r];
    }
    __syncthreads();

    // ---- fused dynamic routing: 256 threads, one (row, dim) each ----
    if (tid < 256) {
        const int row = tid >> 4;
        const int dd  = tid & 15;
        float u[NC];
        #pragma unroll
        for (int n = 0; n < NC; ++n)
            u[n] = lds_u[(row * NC + n) * DC + dd];

        float bl[NC];
        #pragma unroll
        for (int n = 0; n < NC; ++n) bl[n] = 0.f;

        float v = 0.f;
        #pragma unroll
        for (int it = 0; it < 3; ++it) {
            float mx = bl[0];
            #pragma unroll
            for (int n = 1; n < NC; ++n) mx = fmaxf(mx, bl[n]);
            float e[NC], se = 0.f;
            #pragma unroll
            for (int n = 0; n < NC; ++n) { e[n] = __expf(bl[n] - mx); se += e[n]; }
            float inv = 1.f / se;
            float s = 0.f;
            #pragma unroll
            for (int n = 0; n < NC; ++n) s += (e[n] * inv) * u[n];

            float s2 = s * s;
            s2 += __shfl_xor(s2, 1);
            s2 += __shfl_xor(s2, 2);
            s2 += __shfl_xor(s2, 4);
            s2 += __shfl_xor(s2, 8);
            float scale = s2 / ((1.f + s2) * sqrtf(s2 + 1e-7f));
            v = scale * s;

            if (it < 2) {
                #pragma unroll
                for (int n = 0; n < NC; ++n) {
                    float p = u[n] * v;
                    p += __shfl_xor(p, 1);
                    p += __shfl_xor(p, 2);
                    p += __shfl_xor(p, 4);
                    p += __shfl_xor(p, 8);
                    bl[n] += p;
                }
            }
        }
        out[(size_t)(row0 + row) * DC + dd] = v;
    }
}

extern "C" void kernel_launch(void* const* d_in, const int* in_sizes, int n_in,
                              void* d_out, int out_size, void* d_ws, size_t ws_size,
                              hipStream_t stream) {
    const float* inputs = (const float*)d_in[0];   // [B, 2048] fp32
    const float* kernel = (const float*)d_in[1];   // [2048, 160] fp32
    float* out = (float*)d_out;                    // [B, 16] fp32
    const int B = in_sizes[0] / IN_DIM;            // 16384

    unsigned short* wt_hi = (unsigned short*)d_ws;               // [160][2048] bf16
    unsigned short* wt_lo = wt_hi + (size_t)NCOL * IN_DIM;       // [160][2048] bf16

    int nW = IN_DIM * NCOL;
    split_w_kernel<<<dim3((nW + 255) / 256), 256, 0, stream>>>(kernel, wt_hi, wt_lo);
    capsule_fused_kernel<<<dim3(B / 16), 320, 0, stream>>>(inputs, wt_hi, wt_lo, out);
}
